// Encoder_39642548142487
// MI455X (gfx1250) — matrix operand order repaired and checked by reading, not compiled, not run on hardware
//
#include <hip/hip_runtime.h>
#include <hip/hip_bf16.h>
#include <math.h>

#define HDIM 64
#define EPC  60000
#define ETOT (5*EPC)
#define NIN  50000
#define NOUT 50000

typedef __attribute__((ext_vector_type(16))) _Float16 v16h;
typedef __attribute__((ext_vector_type(8)))  float    v8f;

__device__ __forceinline__ unsigned short f2h(float x){
  _Float16 h = (_Float16)x; unsigned short u; __builtin_memcpy(&u,&h,2); return u;
}
__device__ __forceinline__ float h2f(unsigned short u){
  _Float16 h; __builtin_memcpy(&h,&u,2); return (float)h;
}
// Branch-free exact-erf GELU: Abramowitz-Stegun 7.1.26 (|err| < 1.5e-7),
// one v_rcp_f32 + one v_exp_f32 + ~12 FMAs, no EXEC-mask branches.
__device__ __forceinline__ float gelu_f(float x){
  float z = fabsf(x) * 0.70710678118654752f;
  float t = __builtin_amdgcn_rcpf(1.0f + 0.3275911f*z);
  float p = t*(0.254829592f + t*(-0.284496736f + t*(1.421413741f +
            t*(-1.453152027f + t*1.061405429f))));
  float er = 1.0f - p*__expf(-z*z);      // erf(|x|/sqrt(2))
  er = copysignf(er, x);
  return 0.5f*x*(1.0f + er);
}
__device__ __forceinline__ void atomic_max_f32(float* a, float v){
  if (v >= 0.0f) atomicMax((int*)a, __float_as_int(v));
  else atomicMin((unsigned int*)a, (unsigned int)__float_as_int(v));
}

// ---------------------------------------------------------------------------
// WMMA helpers: wave32 v_wmma_f32_16x16x32_f16, D = A(16xK) * B(Kx16) + C.
// A fragment packed from LDS row-major f16 activations per ISA 7.12.2:
//   lane<16: M=lane, halves cover K {0..7,16..23}; lane>=16: M=lane-16, K {8..15,24..31}
//   -> union u32[i] holds K = 2i + (i>=4 ? 8 : 0) (+8 for high lanes), pairs contiguous.
// B fragment is pre-swizzled in global memory: 16 contiguous halves per lane.
// ---------------------------------------------------------------------------
__device__ __forceinline__ v8f wmma_chain(const unsigned short* in, int ldin, int row0,
                                          int ksteps, const unsigned short* wfrag,
                                          int lane, int nt, v8f c){
  int m  = row0 + (lane & 15);
  int hi = (lane >> 4) & 1;
  const unsigned short* pa0 = in + m*ldin + hi*8;
  const unsigned short* pb0 = wfrag + (nt*ksteps*32 + lane)*16;
  for (int ks = 0; ks < ksteps; ++ks){
    union { v16h v; unsigned u[8]; } fa;
    const unsigned short* pa = pa0 + ks*32;
#pragma unroll
    for (int i = 0; i < 8; ++i){
      int k = 2*i + ((i >= 4) ? 8 : 0);
      fa.u[i] = *(const unsigned*)(pa + k);
    }
    v16h b = *(const v16h*)(pb0 + ks*512);
    c = __builtin_amdgcn_wmma_f32_16x16x32_f16(false, fa.v, false, b, (short)0, c,
                                               false, false);
  }
  return c;
}

// C/D layout: vgpr v holds row m = v + 8*(lane>=16), col n = nt*16 + (lane&15)
__device__ __forceinline__ void ep_lds(v8f c, const float* bias, int do_gelu,
                                       unsigned short* out, int row0, int lane, int nt){
  int n  = nt*16 + (lane & 15);
  int mo = ((lane >> 4) & 1)*8;
  float bv = bias[n];
#pragma unroll
  for (int v = 0; v < 8; ++v){
    float x = c[v] + bv;
    if (do_gelu) x = gelu_f(x);
    out[(row0+mo+v)*66 + n] = f2h(x);
  }
}

// ---------------------------------------------------------------------------
// Weight prep: f32 [Din x 64] row-major -> f16 WMMA B fragments (x scale).
// frag[((nt*KS + ks)*32 + lane)*16 + h] = scale*W[(ks*32 + 8*(lane>=16) + h + (h>=8 ? 8 : 0))*64 + nt*16+lane%16]
// ---------------------------------------------------------------------------
__global__ void k_prep_w(const float* __restrict__ W, unsigned short* __restrict__ frag,
                         int KS, float scale){
  int i = blockIdx.x*256 + threadIdx.x;
  int total = 4*KS*32;
  if (i >= total) return;
  int lane = i & 31;
  int ks   = (i >> 5) % KS;
  int nt   = i / (32*KS);
  int n  = nt*16 + (lane & 15);
  int kb = ks*32 + ((lane >> 4) & 1)*8;
  unsigned short* dst = frag + ((size_t)((nt*KS + ks)*32 + lane))*16;
#pragma unroll
  for (int h = 0; h < 16; ++h) dst[h] = f2h(scale * W[(size_t)(kb + h + ((h >= 8) ? 8 : 0))*HDIM + n]);
}

__global__ void k_init(float* mbuf, float* dbuf, float* nfout, float* G){
  size_t i = (size_t)blockIdx.x*256 + threadIdx.x;
  size_t nm = (size_t)(NOUT+1)*HDIM;
  if (i < nm){ mbuf[i] = -__builtin_huge_valf(); dbuf[i] = 0.0f; }
  if (i < (size_t)NOUT*HDIM) nfout[i] = 0.0f;
  if (i < 4096) G[i] = 0.0f;
}

__global__ void k_nfin(const float* __restrict__ volt, const float* __restrict__ W,
                       const float* __restrict__ b, float* __restrict__ nf_in){
  int i = blockIdx.x*256 + threadIdx.x;
  if (i >= NIN*HDIM) return;
  int node = i >> 6, c = i & 63;
  nf_in[i] = volt[node]*W[c] + b[c];
}

// ---------------------------------------------------------------------------
// ef1: per-type 3-layer MLP.  Layer1 (K=1) on VALU, layers 2-3 via WMMA.
// Block: 256 thr / 8 waves = 2 edge-tiles x 4 n-tiles = 32 edges.
// Output staged through LDS -> coalesced dword stores.
// ---------------------------------------------------------------------------
__global__ void k_ef1(const float* __restrict__ dp,
                      const float* __restrict__ W1, const float* __restrict__ b1,
                      const unsigned short* __restrict__ W2f, const float* __restrict__ b2,
                      const unsigned short* __restrict__ W3f, const float* __restrict__ b3,
                      unsigned short* __restrict__ ef1h){
  __shared__ unsigned short sA[32*66];
  __shared__ unsigned short sB[32*66];
  int tid = threadIdx.x;
  int e0  = blockIdx.x*32;
  int t   = e0 / EPC;                       // 60000 % 32 == 0 -> uniform per block
  for (int i = tid; i < 2048; i += 256){
    int el = i >> 6, col = i & 63;
    float x = dp[e0 + el];
    sA[el*66 + col] = f2h(gelu_f(x*W1[t*HDIM + col] + b1[t*HDIM + col]));
  }
  __syncthreads();
  int lane = tid & 31, wave = tid >> 5;
  int row0 = (wave >> 2)*16, nt = wave & 3;
  v8f z = {};
  v8f c = wmma_chain(sA, 66, row0, 2, W2f + (size_t)t*4096, lane, nt, z);
  ep_lds(c, b2 + t*HDIM, 1, sB, row0, lane, nt);
  __syncthreads();
  c = wmma_chain(sB, 66, row0, 2, W3f + (size_t)t*4096, lane, nt, z);
  ep_lds(c, b3 + t*HDIM, 0, sA, row0, lane, nt);
  __syncthreads();
  for (int i = tid; i < 1024; i += 256){
    int el = i >> 5, j = i & 31;
    ((unsigned*)(ef1h + (size_t)(e0 + el)*HDIM))[j] = ((const unsigned*)(sA + el*66))[j];
  }
}

// ---------------------------------------------------------------------------
// Gram matrix G = ef1^T * ef1: stage 32-edge chunk transposed in LDS,
// 16 (mt,nt) output tiles over 8 waves (2 tiles each), atomicAdd into G.
// ---------------------------------------------------------------------------
__device__ __forceinline__ v8f gram_tile(const unsigned short* T, int tile, int lane, v8f c){
  int mt = tile >> 2, nt = tile & 3;
  int hi = (lane >> 4) & 1;
  const unsigned short* pa = T + (mt*16 + (lane & 15))*34 + hi*8;
  const unsigned short* pb = T + (nt*16 + (lane & 15))*34 + hi*8;
  union { v16h v; unsigned u[8]; } fa, fb;
#pragma unroll
  for (int i = 0; i < 8; ++i){
    int ka = 2*i + ((i >= 4) ? 8 : 0);
    fa.u[i] = *(const unsigned*)(pa + ka);
    fb.u[i] = *(const unsigned*)(pb + ka);
  }
  return __builtin_amdgcn_wmma_f32_16x16x32_f16(false, fa.v, false, fb.v, (short)0, c,
                                                false, false);
}
__device__ __forceinline__ void gram_flush(v8f c, int tile, int lane, float* G){
  int mt = tile >> 2, nt = tile & 3;
  int n  = nt*16 + (lane & 15);
  int mo = ((lane >> 4) & 1)*8;
#pragma unroll
  for (int v = 0; v < 8; ++v) atomicAdd(&G[(mt*16 + mo + v)*HDIM + n], c[v]);
}
__global__ void k_gram(const unsigned short* __restrict__ ef1h, float* __restrict__ G){
  __shared__ unsigned short T[64*34];      // transposed chunk: [col][edge]
  int tid = threadIdx.x, lane = tid & 31, wave = tid >> 5;
  v8f a0 = {}, a1 = {};
  for (int ch = blockIdx.x; ch < ETOT/32; ch += gridDim.x){
    int e0 = ch*32;
    __syncthreads();
    for (int i = tid; i < 2048; i += 256){
      int el = i >> 6, c = i & 63;
      T[c*34 + el] = ef1h[(size_t)(e0 + el)*HDIM + c];
    }
    __syncthreads();
    a0 = gram_tile(T, wave,     lane, a0);
    a1 = gram_tile(T, wave + 8, lane, a1);
  }
  gram_flush(a0, wave,     lane, G);
  gram_flush(a1, wave + 8, lane, G);
}

// ---------------------------------------------------------------------------
// Segment softmax passes. logits = ef1 @ G recomputed each pass from the f16
// ef1 chunk in LDS and G fragments pre-scaled by 1/256 (avoids f16 overflow:
// diag(G) ~ 2e4).  Epilogue multiplies the f32 accumulator by 256 to restore
// the exact logits (softmax is not scale invariant).
// mode 0: atomicMax(m[seg,col]); mode 1: atomicAdd(den, exp(l-m));
// mode 2: atomicAdd(nf1_output[seg,col], exp(l-m)/den * ef1).
// ---------------------------------------------------------------------------
__global__ void k_seg(const unsigned short* __restrict__ ef1h,
                      const unsigned short* __restrict__ gfrag,
                      const int* __restrict__ dst_types, const int* __restrict__ dst_nodes,
                      float* __restrict__ mbuf, float* __restrict__ dbuf,
                      float* __restrict__ nfout, int mode){
  __shared__ unsigned short sE[32*66];
  __shared__ int sSeg[32];
  int tid = threadIdx.x;
  int e0  = blockIdx.x*32;
  for (int i = tid; i < 1024; i += 256){
    int el = i >> 5, j = i & 31;
    ((unsigned*)(sE + el*66))[j] = ((const unsigned*)(ef1h + (size_t)(e0 + el)*HDIM))[j];
  }
  if (tid < 32){
    int e = e0 + tid;
    sSeg[tid] = (dst_types[e] == 1) ? dst_nodes[e] : NOUT;
  }
  __syncthreads();
  int lane = tid & 31, wave = tid >> 5;
  int row0 = (wave >> 2)*16, nt = wave & 3;
  v8f z = {};
  v8f c = wmma_chain(sE, 66, row0, 2, gfrag, lane, nt, z);
  int n  = nt*16 + (lane & 15);
  int mo = ((lane >> 4) & 1)*8;
#pragma unroll
  for (int v = 0; v < 8; ++v){
    int ml  = row0 + mo + v;
    int seg = sSeg[ml];
    size_t off = (size_t)seg*HDIM + n;
    float lg = c[v] * 256.0f;              // undo 1/256 G prescale
    if (mode == 0){
      atomic_max_f32(&mbuf[off], lg);
    } else if (mode == 1){
      atomicAdd(&dbuf[off], __expf(lg - mbuf[off]));
    } else if (seg < NOUT){
      float att = __expf(lg - mbuf[off]) / dbuf[off];
      atomicAdd(&nfout[off], att * h2f(sE[ml*66 + n]));
    }
  }
}

// ---------------------------------------------------------------------------
// Fused edge-MLP: gather x=[ef1|src|dst(|vg|vb)] -> 4-layer per-type MLP
// -> + self-MLP(ef1, 4 layers) -> score GEMM -> d_out.  32 edges per block.
// ---------------------------------------------------------------------------
__global__ void __launch_bounds__(256)
k_ef2(int e_start, int nparts, int tbase,
      const unsigned short* __restrict__ ef1h,
      const float* __restrict__ nf_in, const float* __restrict__ nf_out,
      const int* __restrict__ src_t, const int* __restrict__ src_n,
      const int* __restrict__ dst_t, const int* __restrict__ dst_n,
      const int* __restrict__ vg_t,  const int* __restrict__ vg_i,
      const int* __restrict__ vb_t,  const int* __restrict__ vb_i,
      const unsigned short* __restrict__ W1f, const float* __restrict__ b1,
      const unsigned short* __restrict__ W2f, const float* __restrict__ b2,
      const unsigned short* __restrict__ W3f, const float* __restrict__ b3,
      const unsigned short* __restrict__ W4f, const float* __restrict__ b4,
      const unsigned short* __restrict__ sW1f, const float* __restrict__ sb1,
      const unsigned short* __restrict__ sW2f, const float* __restrict__ sb2,
      const unsigned short* __restrict__ sW3f, const float* __restrict__ sb3,
      const unsigned short* __restrict__ sW4f, const float* __restrict__ sb4,
      const unsigned short* __restrict__ scWf, const float* __restrict__ scb,
      float* __restrict__ out){
  __shared__ unsigned short X[32*322];     // concat input, f16, row stride 322
  __shared__ unsigned short bA[32*66];
  __shared__ unsigned short bB[32*66];
  __shared__ unsigned short bC[32*66];
  int tid = threadIdx.x;
  int e0  = e_start + blockIdx.x*32;
  int t   = e0 / EPC;
  int wt  = t - tbase;
  int KS1 = nparts*2;

  // gather: two threads per (edge, part) row; b128 loads, packed f16 stores
  int items = 32*nparts*2;
  for (int r = tid; r < items; r += 256){
    int half = r & 1;
    int row  = r >> 1;
    int el = row / nparts, p = row % nparts;
    int e  = e0 + el;
    unsigned* d = (unsigned*)(X + el*322 + p*64 + half*32);
    if (p == 0){
      const uint4* s = (const uint4*)(ef1h + (size_t)e*HDIM + half*32);
#pragma unroll
      for (int j = 0; j < 4; ++j){
        uint4 q = s[j];
        d[4*j] = q.x; d[4*j+1] = q.y; d[4*j+2] = q.z; d[4*j+3] = q.w;
      }
    } else {
      int ty, ix;
      if      (p == 1){ ty = src_t[e]; ix = src_n[e]; }
      else if (p == 2){ ty = dst_t[e]; ix = dst_n[e]; }
      else if (p == 3){ ty = vg_t[e];  ix = vg_i[e];  }
      else            { ty = vb_t[e];  ix = vb_i[e];  }
      const float4* s = (const float4*)((ty == 0 ? nf_in : nf_out)
                                        + (size_t)ix*HDIM + half*32);
#pragma unroll
      for (int j = 0; j < 8; ++j){
        float4 f = s[j];
        d[2*j]   = (unsigned)f2h(f.x) | ((unsigned)f2h(f.y) << 16);
        d[2*j+1] = (unsigned)f2h(f.z) | ((unsigned)f2h(f.w) << 16);
      }
    }
  }
  __syncthreads();
  int lane = tid & 31, wave = tid >> 5;
  int row0 = (wave >> 2)*16, nt = wave & 3;
  v8f z = {};
  v8f c;
  // edge MLP
  c = wmma_chain(X, 322, row0, KS1, W1f + (size_t)wt*2048*KS1, lane, nt, z);
  ep_lds(c, b1 + wt*HDIM, 1, bA, row0, lane, nt);  __syncthreads();
  c = wmma_chain(bA, 66, row0, 2, W2f + (size_t)wt*4096, lane, nt, z);
  ep_lds(c, b2 + wt*HDIM, 1, bB, row0, lane, nt);  __syncthreads();
  c = wmma_chain(bB, 66, row0, 2, W3f + (size_t)wt*4096, lane, nt, z);
  ep_lds(c, b3 + wt*HDIM, 1, bA, row0, lane, nt);  __syncthreads();
  c = wmma_chain(bA, 66, row0, 2, W4f + (size_t)wt*4096, lane, nt, z);
  ep_lds(c, b4 + wt*HDIM, 0, bC, row0, lane, nt);  __syncthreads();
  // self MLP on ef1 (part 0 of X)
  c = wmma_chain(X, 322, row0, 2, sW1f, lane, nt, z);
  ep_lds(c, sb1, 1, bA, row0, lane, nt);  __syncthreads();
  c = wmma_chain(bA, 66, row0, 2, sW2f, lane, nt, z);
  ep_lds(c, sb2, 1, bB, row0, lane, nt);  __syncthreads();
  c = wmma_chain(bB, 66, row0, 2, sW3f, lane, nt, z);
  ep_lds(c, sb3, 1, bA, row0, lane, nt);  __syncthreads();
  c = wmma_chain(bA, 66, row0, 2, sW4f, lane, nt, z);
  {
    int n  = nt*16 + (lane & 15);
    int mo = ((lane >> 4) & 1)*8;
    float bv = sb4[n];
#pragma unroll
    for (int v = 0; v < 8; ++v){
      float x = c[v] + bv + h2f(bC[(row0 + mo + v)*66 + n]);   // ef2 = mlp + self
      bB[(row0 + mo + v)*66 + n] = f2h(x);
    }
  }
  __syncthreads();
  // score layer -> f32 output (wave covers 16 contiguous cols per row)
  c = wmma_chain(bB, 66, row0, 2, scWf, lane, nt, z);
  {
    int n  = nt*16 + (lane & 15);
    int mo = ((lane >> 4) & 1)*8;
    float bv = scb[n];
#pragma unroll
    for (int v = 0; v < 8; ++v)
      out[(size_t)(e0 + row0 + mo + v)*HDIM + n] = c[v] + bv;
  }
}

// ---------------------------------------------------------------------------
extern "C" void kernel_launch(void* const* d_in, const int* in_sizes, int n_in,
                              void* d_out, int out_size, void* d_ws, size_t ws_size,
                              hipStream_t stream){
  (void)in_sizes; (void)n_in; (void)out_size; (void)ws_size;
  const float* volt  = (const float*)d_in[0];
  const float* dp    = (const float*)d_in[1];
  const int* src_t   = (const int*)d_in[2];
  const int* dst_t   = (const int*)d_in[3];
  const int* src_n   = (const int*)d_in[4];
  const int* dst_n   = (const int*)d_in[5];
  const int* vg_t    = (const int*)d_in[6];
  const int* vg_i    = (const int*)d_in[7];
  const int* vb_t    = (const int*)d_in[8];
  const int* vb_i    = (const int*)d_in[9];
  const float* nf1W  = (const float*)d_in[11];
  const float* nf1b  = (const float*)d_in[12];
  const float* e1W1  = (const float*)d_in[13]; const float* e1b1 = (const float*)d_in[14];
  const float* e1W2  = (const float*)d_in[15]; const float* e1b2 = (const float*)d_in[16];
  const float* e1W3  = (const float*)d_in[17]; const float* e1b3 = (const float*)d_in[18];
  const float* mW1   = (const float*)d_in[19]; const float* mb1  = (const float*)d_in[20];
  const float* mW2   = (const float*)d_in[21]; const float* mb2  = (const float*)d_in[22];
  const float* mW3   = (const float*)d_in[23]; const float* mb3  = (const float*)d_in[24];
  const float* mW4   = (const float*)d_in[25]; const float* mb4  = (const float*)d_in[26];
  const float* rW1   = (const float*)d_in[27]; const float* rb1  = (const float*)d_in[28];
  const float* rW2   = (const float*)d_in[29]; const float* rb2  = (const float*)d_in[30];
  const float* rW3   = (const float*)d_in[31]; const float* rb3  = (const float*)d_in[32];
  const float* rW4   = (const float*)d_in[33]; const float* rb4  = (const float*)d_in[34];
  const float* sW1   = (const float*)d_in[35]; const float* sb1  = (const float*)d_in[36];
  const float* sW2   = (const float*)d_in[37]; const float* sb2  = (const float*)d_in[38];
  const float* sW3   = (const float*)d_in[39]; const float* sb3  = (const float*)d_in[40];
  const float* sW4   = (const float*)d_in[41]; const float* sb4  = (const float*)d_in[42];
  const float* scW   = (const float*)d_in[43]; const float* scb  = (const float*)d_in[44];
  float* out = (float*)d_out;

  char* base = (char*)d_ws;
  size_t off = 0;
  auto alloc = [&](size_t bytes) -> void* {
    void* p = base + off; off = (off + bytes + 255) & ~(size_t)255; return p;
  };
  unsigned short* ef1h = (unsigned short*)alloc((size_t)ETOT*HDIM*2);
  float* nfin  = (float*)alloc((size_t)NIN *HDIM*4);
  float* nfout = (float*)alloc((size_t)NOUT*HDIM*4);
  float* mbuf  = (float*)alloc((size_t)(NOUT+1)*HDIM*4);
  float* dbuf  = (float*)alloc((size_t)(NOUT+1)*HDIM*4);
  float* G     = (float*)alloc(4096*4);
  unsigned short* gF = (unsigned short*)alloc(4096*2);
  auto fhal = [](int ks){ return (size_t)2048*ks; };        // frag halves per type
  unsigned short* e1W2f = (unsigned short*)alloc(5*fhal(2)*2);
  unsigned short* e1W3f = (unsigned short*)alloc(5*fhal(2)*2);
  unsigned short* mW1f  = (unsigned short*)alloc(2*fhal(10)*2);
  unsigned short* mW2f  = (unsigned short*)alloc(2*fhal(2)*2);
  unsigned short* mW3f  = (unsigned short*)alloc(2*fhal(2)*2);
  unsigned short* mW4f  = (unsigned short*)alloc(2*fhal(2)*2);
  unsigned short* rW1f  = (unsigned short*)alloc(3*fhal(6)*2);
  unsigned short* rW2f  = (unsigned short*)alloc(3*fhal(2)*2);
  unsigned short* rW3f  = (unsigned short*)alloc(3*fhal(2)*2);
  unsigned short* rW4f  = (unsigned short*)alloc(3*fhal(2)*2);
  unsigned short* sW1f  = (unsigned short*)alloc(fhal(2)*2);
  unsigned short* sW2f  = (unsigned short*)alloc(fhal(2)*2);
  unsigned short* sW3f  = (unsigned short*)alloc(fhal(2)*2);
  unsigned short* sW4f  = (unsigned short*)alloc(fhal(2)*2);
  unsigned short* scWf  = (unsigned short*)alloc(fhal(2)*2);

  auto prep = [&](const float* W, unsigned short* f, int KS, float scale){
    int tot = 4*KS*32;
    k_prep_w<<<dim3((tot + 255)/256), dim3(256), 0, stream>>>(W, f, KS, scale);
  };

  // init accumulators
  {
    size_t nm = (size_t)(NOUT+1)*HDIM;
    k_init<<<dim3((unsigned)((nm + 255)/256)), dim3(256), 0, stream>>>(mbuf, dbuf, nfout, G);
  }
  // weight fragment prep
  for (int t = 0; t < 5; ++t){ prep(e1W2 + (size_t)t*4096, e1W2f + t*fhal(2), 2, 1.0f);
                               prep(e1W3 + (size_t)t*4096, e1W3f + t*fhal(2), 2, 1.0f); }
  for (int t = 0; t < 2; ++t){ prep(mW1 + (size_t)t*320*64, mW1f + t*fhal(10), 10, 1.0f);
                               prep(mW2 + (size_t)t*4096,   mW2f + t*fhal(2),  2, 1.0f);
                               prep(mW3 + (size_t)t*4096,   mW3f + t*fhal(2),  2, 1.0f);
                               prep(mW4 + (size_t)t*4096,   mW4f + t*fhal(2),  2, 1.0f); }
  for (int t = 0; t < 3; ++t){ prep(rW1 + (size_t)t*192*64, rW1f + t*fhal(6), 6, 1.0f);
                               prep(rW2 + (size_t)t*4096,   rW2f + t*fhal(2), 2, 1.0f);
                               prep(rW3 + (size_t)t*4096,   rW3f + t*fhal(2), 2, 1.0f);
                               prep(rW4 + (size_t)t*4096,   rW4f + t*fhal(2), 2, 1.0f); }
  prep(sW1, sW1f, 2, 1.0f); prep(sW2, sW2f, 2, 1.0f);
  prep(sW3, sW3f, 2, 1.0f); prep(sW4, sW4f, 2, 1.0f);
  prep(scW, scWf, 2, 1.0f);

  // node features for 'input' table
  k_nfin<<<dim3((NIN*HDIM + 255)/256), dim3(256), 0, stream>>>(volt, nf1W, nf1b, nfin);
  // ef1 (per-type 3-layer MLP)
  k_ef1<<<dim3(ETOT/32), dim3(256), 0, stream>>>(dp, e1W1, e1b1, e1W2f, e1b2,
                                                 e1W3f, e1b3, ef1h);
  // G = ef1^T ef1, then swizzle to B-fragments with 1/256 prescale (f16 range)
  k_gram<<<dim3(512), dim3(256), 0, stream>>>(ef1h, G);
  prep(G, gF, 2, 1.0f/256.0f);
  // segment softmax: max, denom, weighted scatter-sum
  k_seg<<<dim3(ETOT/32), dim3(256), 0, stream>>>(ef1h, gF, dst_t, dst_n, mbuf, dbuf, nfout, 0);
  k_seg<<<dim3(ETOT/32), dim3(256), 0, stream>>>(ef1h, gF, dst_t, dst_n, mbuf, dbuf, nfout, 1);
  k_seg<<<dim3(ETOT/32), dim3(256), 0, stream>>>(ef1h, gF, dst_t, dst_n, mbuf, dbuf, nfout, 2);
  // fused edge MLP + self MLP + score
  k_ef2<<<dim3(2*EPC/32), dim3(256), 0, stream>>>(0, 5, 0, ef1h, nfin, nfout,
      src_t, src_n, dst_t, dst_n, vg_t, vg_i, vb_t, vb_i,
      mW1f, mb1, mW2f, mb2, mW3f, mb3, mW4f, mb4,
      sW1f, sb1, sW2f, sb2, sW3f, sb3, sW4f, sb4, scWf, scb, out);
  k_ef2<<<dim3(3*EPC/32), dim3(256), 0, stream>>>(2*EPC, 3, 2, ef1h, nfin, nfout,
      src_t, src_n, dst_t, dst_n, vg_t, vg_i, vb_t, vb_i,
      rW1f, rb1, rW2f, rb2, rW3f, rb3, rW4f, rb4,
      sW1f, sb1, sW2f, sb2, sW3f, sb3, sW4f, sb4, scWf, scb, out);
}
